// SSAttn_28329604284591
// MI455X (gfx1250) — matrix operand order repaired and checked by reading, not compiled, not run on hardware
//
#include <hip/hip_runtime.h>
#include <hip/hip_bf16.h>

// ---------------------------------------------------------------------------
// Fused self-attention for [B=2,H=12,S=4096,D=64], HID=64, fp32 in/out.
// Compute-bound (~107 GFLOP vs ~50MB HBM): all MACs on v_wmma_f32_16x16x32_f16.
// Kernel 1 builds f16 Q (pre-scaled by 1/sqrt(64)), K, V^T in d_ws.
// Kernel 2: flash attention. Each block (8 waves) shares async-to-LDS staged
// 64-key K/V chunks (double buffered, ASYNCcnt), each wave owns 16 query rows
// with online softmax in registers; output projection fused in the epilogue.
// ---------------------------------------------------------------------------

typedef __attribute__((ext_vector_type(16))) _Float16 v16h;
typedef __attribute__((ext_vector_type(8)))  float    v8f;
typedef __attribute__((ext_vector_type(4)))  float    v4f;

#define S_LEN 4096
#define DIMC  64
#define BH    24
#define KCHUNK 64                    // keys per staged chunk

__device__ __forceinline__ v8f wmma16(v16h a, v16h b, v8f c) {
    return __builtin_amdgcn_wmma_f32_16x16x32_f16(false, a, false, b,
                                                  (short)0, c, false, false);
}

// A-operand (16x32 f16, MxK): lane holds row M=lane&15, 16 contiguous K
// values at kb=(lane>>4)*16 (matches B's contiguous K order).
__device__ __forceinline__ v16h loadA16(const _Float16* base, int ld, int lane) {
    int m = lane & 15, kb = (lane >> 4) << 4;
    const _Float16* p = base + m * ld + kb;
    union { v16h h; v4f f[2]; } u;
    u.f[0] = *(const v4f*)(p);
    u.f[1] = *(const v4f*)(p + 8);
    return u.h;
}

// Same A layout but converting from fp32 source.
__device__ __forceinline__ v16h loadA32cvt(const float* base, int ld, int lane, int koff) {
    int m = lane & 15, kb = ((lane >> 4) << 4) + koff;
    const float* p = base + m * ld + kb;
    v4f f0 = ((const v4f*)p)[0];
    v4f f1 = ((const v4f*)p)[1];
    v4f f2 = ((const v4f*)p)[2];
    v4f f3 = ((const v4f*)p)[3];
    v16h h;
#pragma unroll
    for (int i = 0; i < 4; ++i) {
        h[i]      = (_Float16)f0[i];
        h[4 + i]  = (_Float16)f1[i];
        h[8 + i]  = (_Float16)f2[i];
        h[12 + i] = (_Float16)f3[i];
    }
    return h;
}

// B-operand (32x16 f16, KxN): lane holds column N=lane&15 with 16 contiguous
// K values; caller passes the per-lane base pointer (already includes the
// (lane>>4)*16 K offset). 32B contiguous load (global or DS).
__device__ __forceinline__ v16h loadB16(const _Float16* p) {
    union { v16h h; v4f f[2]; } u;
    u.f[0] = *(const v4f*)(p);
    u.f[1] = *(const v4f*)(p + 8);
    return u.h;
}

// B-operand converting from fp32 source (for weight matrices).
__device__ __forceinline__ v16h loadB32cvt(const float* p) {
    v4f f0 = ((const v4f*)p)[0];
    v4f f1 = ((const v4f*)p)[1];
    v4f f2 = ((const v4f*)p)[2];
    v4f f3 = ((const v4f*)p)[3];
    v16h h;
#pragma unroll
    for (int i = 0; i < 4; ++i) {
        h[i]      = (_Float16)f0[i];
        h[4 + i]  = (_Float16)f1[i];
        h[8 + i]  = (_Float16)f2[i];
        h[12 + i] = (_Float16)f3[i];
    }
    return h;
}

// max/sum reduction across the 16-lane half holding one C-layout row.
__device__ __forceinline__ float half16_max(float v) {
#pragma unroll
    for (int m = 1; m <= 8; m <<= 1) v = fmaxf(v, __shfl_xor(v, m, 32));
    return v;
}
__device__ __forceinline__ float half16_sum(float v) {
#pragma unroll
    for (int m = 1; m <= 8; m <<= 1) v += __shfl_xor(v, m, 32);
    return v;
}

// Low 32 bits of a generic pointer into __shared__ = LDS byte offset.
__device__ __forceinline__ unsigned lds_of(const void* p) {
    return (unsigned)(unsigned long long)p;
}

// 16B async copy global -> LDS (CDNA5 VGLOBAL opcode 98, ASYNCcnt-tracked).
__device__ __forceinline__ void async_b128(unsigned lds_off, const void* gptr) {
    unsigned long long g = (unsigned long long)gptr;
    asm volatile("global_load_async_to_lds_b128 %0, %1, off"
                 :: "v"(lds_off), "v"(g) : "memory");
}

// ---------------------------------------------------------------------------
// Kernel 1: q = (x Wq^T + bq) / sqrt(64), k = x Wk^T + bk   (f16 [s][e])
//           vT = (x Wv^T + bv)^T                            (f16 [e][s])
// ---------------------------------------------------------------------------
__global__ void ssattn_qkv_kernel(const float* __restrict__ x,
                                  const float* __restrict__ Wq, const float* __restrict__ bq,
                                  const float* __restrict__ Wk, const float* __restrict__ bk,
                                  const float* __restrict__ Wv, const float* __restrict__ bv,
                                  _Float16* __restrict__ q, _Float16* __restrict__ k,
                                  _Float16* __restrict__ vT) {
    const int bh    = blockIdx.x >> 5;
    const int stile = blockIdx.x & 31;
    const int wave  = threadIdx.x >> 5;
    const int lane  = threadIdx.x & 31;
    const int col   = lane & 15;
    const int hi16  = (lane >> 4) << 4;
    const int hi8   = (lane >> 4) << 3;
    const int srow0 = stile * 128 + wave * 16;

    const float* xp = x + ((size_t)bh * S_LEN + srow0) * DIMC;
    const v16h a0 = loadA32cvt(xp, DIMC, lane, 0);    // K = d 0..31
    const v16h a1 = loadA32cvt(xp, DIMC, lane, 32);   // K = d 32..63

    const float* Ws[3]  = {Wq, Wk, Wv};
    const float* bs[3]  = {bq, bk, bv};
    const float  scl[3] = {0.125f, 1.0f, 1.0f};       // fold 1/sqrt(64) into q

#pragma unroll
    for (int w = 0; w < 3; ++w) {
#pragma unroll
        for (int nt = 0; nt < 4; ++nt) {
            const float* wrow = Ws[w] + (size_t)(nt * 16 + col) * DIMC + hi16;
            v8f acc = {};
            acc = wmma16(a0, loadB32cvt(wrow), acc);
            acc = wmma16(a1, loadB32cvt(wrow + 32), acc);
            const float bias = bs[w][nt * 16 + col];
#pragma unroll
            for (int r = 0; r < 8; ++r) {
                const float val = (acc[r] + bias) * scl[w];
                const int   row = r + hi8;                       // C-layout row
                const int   e   = nt * 16 + col;
                if (w == 0)
                    q[((size_t)bh * S_LEN + srow0 + row) * DIMC + e] = (_Float16)val;
                else if (w == 1)
                    k[((size_t)bh * S_LEN + srow0 + row) * DIMC + e] = (_Float16)val;
                else
                    vT[((size_t)bh * DIMC + e) * S_LEN + srow0 + row] = (_Float16)val;
            }
        }
    }
}

// ---------------------------------------------------------------------------
// Kernel 2: flash attention with block-shared async-staged K/V chunks.
// ---------------------------------------------------------------------------
__global__ void ssattn_fa_kernel(const _Float16* __restrict__ q,
                                 const _Float16* __restrict__ k,
                                 const _Float16* __restrict__ vT,
                                 const float* __restrict__ Wp,
                                 const float* __restrict__ bp,
                                 float* __restrict__ out) {
    __shared__ _Float16 kbuf[2][KCHUNK * DIMC];   // 2 x 8KB: 64 key rows x 64 d
    __shared__ _Float16 vbuf[2][DIMC * KCHUNK];   // 2 x 8KB: 64 e rows x 64 keys
    __shared__ _Float16 pstage[8][16 * 64];       // per-wave P/O staging (16KB)

    const int tid   = threadIdx.x;
    const int bh    = blockIdx.x >> 5;
    const int qtile = blockIdx.x & 31;
    const int wave  = tid >> 5;
    const int lane  = tid & 31;
    const int col   = lane & 15;
    const int hi16  = (lane >> 4) << 4;
    const int hi8   = (lane >> 4) << 3;
    const int q0    = qtile * 128 + wave * 16;

    _Float16* pb = pstage[wave];

    const _Float16* qbh = q  + (size_t)bh * S_LEN * DIMC;
    const _Float16* kbh = k  + (size_t)bh * S_LEN * DIMC;
    const _Float16* vbh = vT + (size_t)bh * DIMC * S_LEN;

    // Q tile resident as two A-operands (K-dim 64 = 2 x 32).
    const v16h qa0 = loadA16(qbh + (size_t)q0 * DIMC,      DIMC, lane);
    const v16h qa1 = loadA16(qbh + (size_t)q0 * DIMC + 32, DIMC, lane);

    v8f   o[4] = {v8f{}, v8f{}, v8f{}, v8f{}};    // O accum: 16q x 64e fp32
    float rm[8], rs[8];
#pragma unroll
    for (int r = 0; r < 8; ++r) { rm[r] = -1e30f; rs[r] = 0.0f; }

    // Cooperative async stage of one 64-key chunk (K: 8KB contiguous;
    // V^T: 64 rows x 128B). 4 async B128 per thread, ASYNCcnt-tracked.
    auto stage = [&](int j0, int buf) {
        {   // K chunk: contiguous 8KB, 32B per thread
            const char* g = (const char*)(kbh + (size_t)j0 * DIMC) + tid * 32;
            unsigned    l = lds_of(&kbuf[buf][0]) + tid * 32;
            async_b128(l, g);
            async_b128(l + 16, g + 16);
        }
        {   // V^T chunk: 64 strided 128B rows, 4 threads per row
            const int e = tid >> 2, oo = (tid & 3) * 32;   // byte offset in row
            const char* g = (const char*)(vbh + (size_t)e * S_LEN + j0) + oo;
            unsigned    l = lds_of(&vbuf[buf][0]) + e * 128 + oo;
            async_b128(l, g);
            async_b128(l + 16, g + 16);
        }
    };

    stage(0, 0);

    for (int jc = 0; jc < S_LEN / KCHUNK; ++jc) {
        asm volatile("s_wait_asynccnt 0x0" ::: "memory"); // this chunk landed
        __syncthreads();                                  // ...for all waves
        if (jc + 1 < S_LEN / KCHUNK)
            stage((jc + 1) * KCHUNK, (jc + 1) & 1);       // prefetch next chunk

        const _Float16* kc = kbuf[jc & 1];
        const _Float16* vc = vbuf[jc & 1];

        // ---- S tiles: four 16x16 chunks of Q K^T (scale pre-folded into q)
        v8f st[4];
#pragma unroll
        for (int t = 0; t < 4; ++t) {
            const _Float16* krow = kc + (t * 16 + col) * DIMC + hi16;
            v8f z = {};
            z     = wmma16(qa0, loadB16(krow),      z);
            st[t] = wmma16(qa1, loadB16(krow + 32), z);
        }

        // ---- online softmax state update (per C-layout row)
        float alpha[8];
#pragma unroll
        for (int r = 0; r < 8; ++r) {
            float v    = fmaxf(fmaxf(st[0][r], st[1][r]), fmaxf(st[2][r], st[3][r]));
            float mnew = fmaxf(rm[r], half16_max(v));
            alpha[r]   = __expf(rm[r] - mnew);
            rm[r]      = mnew;
            rs[r]     *= alpha[r];
        }
#pragma unroll
        for (int et = 0; et < 4; ++et)
#pragma unroll
            for (int r = 0; r < 8; ++r) o[et][r] *= alpha[r];

        // ---- P = exp(S - m): row sums + stage f16 tile (16x64) in LDS
#pragma unroll
        for (int r = 0; r < 8; ++r) {
            const int row = r + hi8;
            float ps = 0.0f;
#pragma unroll
            for (int t = 0; t < 4; ++t) {
                const float p = __expf(st[t][r] - rm[r]);
                pb[row * 64 + t * 16 + col] = (_Float16)p;
                ps += p;
            }
            rs[r] += half16_sum(ps);
        }
        // DS ops are wave-in-order; wait for stores before cross-lane reload.
        asm volatile("s_wait_dscnt 0x0" ::: "memory");

        // ---- O += P V : P reloaded in A layout (two 16x32), V^T from LDS
        const v16h pa0 = loadA16(pb,      64, lane);      // keys j0..j0+31
        const v16h pa1 = loadA16(pb + 32, 64, lane);      // keys j0+32..j0+63
#pragma unroll
        for (int et = 0; et < 4; ++et) {
            const _Float16* vrow = vc + (et * 16 + col) * KCHUNK;
            o[et] = wmma16(pa0, loadB16(vrow + hi16),      o[et]);
            o[et] = wmma16(pa1, loadB16(vrow + 32 + hi16), o[et]);
        }
        __syncthreads();   // all waves done with kc/vc before re-stage
    }

    // ---- epilogue: O /= rowsum, bounce to A layout, apply Wp + bp
    float inv[8];
#pragma unroll
    for (int r = 0; r < 8; ++r) inv[r] = 1.0f / rs[r];
#pragma unroll
    for (int et = 0; et < 4; ++et)
#pragma unroll
        for (int r = 0; r < 8; ++r) {
            const int row = r + hi8;
            pb[row * 64 + et * 16 + col] = (_Float16)(o[et][r] * inv[r]);
        }
    asm volatile("s_wait_dscnt 0x0" ::: "memory");

    const v16h oa0 = loadA16(pb,      64, lane);   // e 0..31
    const v16h oa1 = loadA16(pb + 32, 64, lane);   // e 32..63

#pragma unroll
    for (int dt = 0; dt < 4; ++dt) {
        const float* wrow = Wp + (size_t)(dt * 16 + col) * DIMC + hi16;
        v8f c = {};
        c = wmma16(oa0, loadB32cvt(wrow),      c);
        c = wmma16(oa1, loadB32cvt(wrow + 32), c);
        const float bias = bp[dt * 16 + col];
#pragma unroll
        for (int r = 0; r < 8; ++r) {
            const int row = r + hi8;
            out[((size_t)bh * S_LEN + q0 + row) * DIMC + dt * 16 + col] = c[r] + bias;
        }
    }
}

extern "C" void kernel_launch(void* const* d_in, const int* in_sizes, int n_in,
                              void* d_out, int out_size, void* d_ws, size_t ws_size,
                              hipStream_t stream) {
    const float* x  = (const float*)d_in[0];
    const float* Wq = (const float*)d_in[1];
    const float* bq = (const float*)d_in[2];
    const float* Wk = (const float*)d_in[3];
    const float* bk = (const float*)d_in[4];
    const float* Wv = (const float*)d_in[5];
    const float* bv = (const float*)d_in[6];
    const float* Wp = (const float*)d_in[7];
    const float* bp = (const float*)d_in[8];

    const size_t per = (size_t)BH * S_LEN * DIMC;   // 6.29M f16 elems each
    _Float16* qws = (_Float16*)d_ws;                // needs 3*per*2 = 37.7MB ws
    _Float16* kws = qws + per;
    _Float16* vws = kws + per;

    dim3 grid(BH * (S_LEN / 128));                  // 768 blocks
    dim3 block(256);                                // 8 wave32

    ssattn_qkv_kernel<<<grid, block, 0, stream>>>(x, Wq, bq, Wk, bk, Wv, bv,
                                                  qws, kws, vws);
    ssattn_fa_kernel<<<grid, block, 0, stream>>>(qws, kws, vws, Wp, bp,
                                                 (float*)d_out);
}
